// CenteredLSTM_75127567941756
// MI455X (gfx1250) — matrix operand order repaired and checked by reading, not compiled, not run on hardware
//
#include <hip/hip_runtime.h>
#include <hip/hip_bf16.h>

// ---------------------------------------------------------------------------
// CenteredLSTM for MI455X (gfx1250, wave32, WMMA bf16 16x16x32)
// ---------------------------------------------------------------------------
#define T_LEN 256
#define BATCH 64
#define EDIM  512
#define HDIM  512
#define G4H   2048          // 4*H gate dimension
#define TB    (T_LEN*BATCH) // 16384 rows for the big GEMMs

typedef __bf16 bf16;
typedef __attribute__((ext_vector_type(16))) __bf16 bf16x16;
typedef __attribute__((ext_vector_type(8)))  float  f32x8;
typedef __attribute__((ext_vector_type(4)))  unsigned int u32x4;

// A 16x32 / B 32x16 bf16 fragment = 16 bf16 per lane = 32 bytes = 2 x b128.
union Frag { bf16x16 v; u32x4 q[2]; };

__device__ __forceinline__ f32x8 wmma_bf16(const Frag& a, const Frag& b, f32x8 c) {
  // D = A(16x32) * B(32x16) + C, fp32 accumulate
  return __builtin_amdgcn_wmma_f32_16x16x32_bf16(false, a.v, false, b.v,
                                                 (short)0, c, false, false);
}

__device__ __forceinline__ float sigf(float x) { return 1.0f / (1.0f + __expf(-x)); }

// ---------------------------------------------------------------------------
// fp32 -> bf16 conversion (weights)
// ---------------------------------------------------------------------------
__global__ void cvt_k(const float* __restrict__ s, bf16* __restrict__ d, int n) {
  for (int i = blockIdx.x * blockDim.x + threadIdx.x; i < n;
       i += gridDim.x * blockDim.x)
    d[i] = (bf16)s[i];
}

// lens_pre[b] = offsets[b]; lens_fol[b] = max(stc[b]-off[b]-1, 0)
__global__ void prep_lens_k(const int* __restrict__ off, const int* __restrict__ stc,
                            int* __restrict__ lp, int* __restrict__ lf) {
  int b = threadIdx.x;
  if (b < BATCH) {
    lp[b] = off[b];
    int f = stc[b] - off[b] - 1;
    lf[b] = f > 0 ? f : 0;
  }
}

// Build masked (pre) or shifted+masked (fol) input in bf16, layout [T][B][E].
__global__ void prep_x_k(const float* __restrict__ emb, const int* __restrict__ off,
                         const int* __restrict__ stc, bf16* __restrict__ x, int mode) {
  const int n = TB * EDIM;
  for (int i = blockIdx.x * blockDim.x + threadIdx.x; i < n;
       i += gridDim.x * blockDim.x) {
    const int e = i & (EDIM - 1);
    const int b = (i >> 9) & (BATCH - 1);
    const int t = i >> 15;
    float v;
    if (mode == 0) {                    // pre: mask t < offsets[b]
      v = (t < off[b]) ? emb[i] : 0.0f;
    } else {                            // fol: shift by off+1, mask t < fol_len
      int fl = stc[b] - off[b] - 1; if (fl < 0) fl = 0;
      int st = t + off[b] + 1; if (st > T_LEN - 1) st = T_LEN - 1;
      v = (t < fl) ? emb[(((size_t)st * BATCH + b) << 9) + e] : 0.0f;
    }
    x[i] = (bf16)v;
  }
}

// ---------------------------------------------------------------------------
// zx GEMM:  Z[M=TB][2048](bf16) = X[M][K](bf16) * W[2048][K](bf16)^T + bias
// Block: 256 threads = 8 waves; each wave computes a 16x64 strip.
// A-frag (16-bit A 16x32): lane m=L&15, half=L>>4 holds
//   k in [half*16, half*16+16)  -> two contiguous b128 loads.
// B-frag (32x16): lane n=L&15, half=L>>4 holds k in [half*16, half*16+16)
//   -> two contiguous b128 loads from weight row n (row-major [g][k]).
// ---------------------------------------------------------------------------
__global__ __launch_bounds__(256) void zx_gemm_k(
    const bf16* __restrict__ X, const bf16* __restrict__ W,
    const float* __restrict__ bias, bf16* __restrict__ Z, int K) {
  const int lane = threadIdx.x & 31;
  const int wave = threadIdx.x >> 5;
  const int half = lane >> 4;
  const int nl   = lane & 15;
  const int m0 = (blockIdx.y * 8 + wave) * 16;
  const int n0 = blockIdx.x * 64;
  const int kTiles = K >> 5;

  const bf16* Arow = X + (size_t)(m0 + nl) * K;
  const bf16* Wrow = W + (size_t)(n0 + nl) * K;

  f32x8 acc[4] = {};
#pragma unroll 4
  for (int kt = 0; kt < kTiles; ++kt) {
    const int ka = kt * 32 + half * 16;
    Frag a;
    a.q[0] = *(const u32x4*)(Arow + ka);
    a.q[1] = *(const u32x4*)(Arow + ka + 8);
    const int kb = kt * 32 + half * 16;
#pragma unroll
    for (int sub = 0; sub < 4; ++sub) {
      const bf16* wp = Wrow + (size_t)(sub * 16) * K + kb;
      Frag b;
      b.q[0] = *(const u32x4*)(wp);
      b.q[1] = *(const u32x4*)(wp + 8);
      acc[sub] = wmma_bf16(a, b, acc[sub]);
    }
  }
  // C/D layout: lane col = L&15, VGPR v -> row v + 8*(L>>4)
#pragma unroll
  for (int sub = 0; sub < 4; ++sub) {
    const int col = n0 + sub * 16 + nl;
    const float bv = bias[col];
#pragma unroll
    for (int v = 0; v < 8; ++v) {
      const int row = m0 + v + (half << 3);
      Z[(size_t)row * G4H + col] = (bf16)(acc[sub][v] + bv);
    }
  }
}

// ---------------------------------------------------------------------------
// Fused persistent LSTM recurrence: grid.x = nruns (<=4); each block (1024
// threads = 32 waves on one WGP) runs all 256 timesteps of one independent
// (model, direction) run.  h (bf16) lives in LDS and feeds WMMA A fragments;
// c stays in fp32 registers; Whh (bf16, 2MB/run) streams from the 192MB L2.
// Wave w: rows 16*(w&3)..+16  x  h-cols (w>>2)*64..+64 (4 gate tiles per sub).
// Sequence reversal handled by indexing zx / outSeq at reversed time rt(b).
// ---------------------------------------------------------------------------
struct RecArgs {
  const bf16* zx[4];      // [T][B][2048] bf16 gate pre-activations
  const bf16* whh[4];     // [2048][512] bf16
  const int*  lens[4];    // [B]
  bf16*       outSeq[4];  // [T][B][1024] or null
  float*      hFinal[4];  // [B][512] or null
  int         outCol[4];  // 0 (fwd) or 512 (bwd)
  int         reverse[4];
};

__global__ __launch_bounds__(1024) void lstm_rec_k(RecArgs A) {
  const int run = blockIdx.x;
  const bf16* __restrict__ zxp  = A.zx[run];
  const bf16* __restrict__ Whh  = A.whh[run];
  const int*  __restrict__ lens = A.lens[run];
  bf16*       __restrict__ outSeq = A.outSeq[run];
  float*      __restrict__ hFinal = A.hFinal[run];
  const int outColBase = A.outCol[run];
  const int reverse    = A.reverse[run];

  __shared__ __align__(16) bf16 hbuf[BATCH * HDIM];   // 64KB
  const int tid  = threadIdx.x;
  const int lane = tid & 31, wave = tid >> 5;
  const int half = lane >> 4, nl = lane & 15;
  const int mtile = wave & 3;
  const int hblk  = (wave >> 2) * 64;

  for (int i = tid; i < BATCH * HDIM; i += 1024) hbuf[i] = (bf16)0.0f;

  int bIdx[8], bLen[8];
#pragma unroll
  for (int v = 0; v < 8; ++v) {
    bIdx[v] = 16 * mtile + v + (half << 3);   // batch row of accum VGPR v
    bLen[v] = lens[bIdx[v]];
  }
  f32x8 cacc[4] = {};
  __syncthreads();

  for (int t = 0; t < T_LEN; ++t) {
    bf16 hkeep[4][8];
#pragma unroll
    for (int sub = 0; sub < 4; ++sub) {
      const int col = hblk + sub * 16 + nl;   // h column 0..511
      const bf16* Arow = hbuf + (16 * mtile + nl) * HDIM;
      const bf16* W0 = Whh + (size_t)col * HDIM;           // gate i
      const bf16* W1 = W0 + (size_t)HDIM * HDIM;           // gate f
      const bf16* W2 = W1 + (size_t)HDIM * HDIM;           // gate g
      const bf16* W3 = W2 + (size_t)HDIM * HDIM;           // gate o
      f32x8 zi = {}, zf = {}, zg = {}, zo = {};
#pragma unroll 4
      for (int kt = 0; kt < HDIM / 32; ++kt) {
        const int ka = kt * 32 + half * 16;
        Frag a;
        a.q[0] = *(const u32x4*)(Arow + ka);
        a.q[1] = *(const u32x4*)(Arow + ka + 8);
        const int kb = kt * 32 + half * 16;
        Frag b;
        b.q[0] = *(const u32x4*)(W0 + kb); b.q[1] = *(const u32x4*)(W0 + kb + 8);
        zi = wmma_bf16(a, b, zi);
        b.q[0] = *(const u32x4*)(W1 + kb); b.q[1] = *(const u32x4*)(W1 + kb + 8);
        zf = wmma_bf16(a, b, zf);
        b.q[0] = *(const u32x4*)(W2 + kb); b.q[1] = *(const u32x4*)(W2 + kb + 8);
        zg = wmma_bf16(a, b, zg);
        b.q[0] = *(const u32x4*)(W3 + kb); b.q[1] = *(const u32x4*)(W3 + kb + 8);
        zo = wmma_bf16(a, b, zo);
      }
#pragma unroll
      for (int v = 0; v < 8; ++v) {
        const int b   = bIdx[v];
        const int len = bLen[v];
        const int tt  = reverse ? ((t < len) ? (len - 1 - t) : t) : t;
        const size_t zrow = ((size_t)tt * BATCH + b) * G4H;
        const float ziv = zi[v] + (float)zxp[zrow + col];
        const float zfv = zf[v] + (float)zxp[zrow + 512 + col];
        const float zgv = zg[v] + (float)zxp[zrow + 1024 + col];
        const float zov = zo[v] + (float)zxp[zrow + 1536 + col];
        // Prefetch next timestep's zx gate rows into cache (global_prefetch).
        if (t + 1 < T_LEN) {
          const int tn = t + 1;
          const int ttn = reverse ? ((tn < len) ? (len - 1 - tn) : tn) : tn;
          const size_t zr2 = ((size_t)ttn * BATCH + b) * G4H;
          __builtin_prefetch(&zxp[zr2 + col], 0, 3);
          __builtin_prefetch(&zxp[zr2 + 512 + col], 0, 3);
          __builtin_prefetch(&zxp[zr2 + 1024 + col], 0, 3);
          __builtin_prefetch(&zxp[zr2 + 1536 + col], 0, 3);
        }
        const float ig = sigf(ziv), fg = sigf(zfv), og = sigf(zov);
        const float gg = tanhf(zgv);
        const float cOld = cacc[sub][v];
        const float cNew = fg * cOld + ig * gg;
        const float hNew = og * tanhf(cNew);
        const bool  m    = (t < len);
        const float hOld = (float)hbuf[b * HDIM + col];
        cacc[sub][v] = m ? cNew : cOld;
        hkeep[sub][v] = (bf16)(m ? hNew : hOld);
        if (outSeq) {
          outSeq[((size_t)tt * BATCH + b) * (2 * HDIM) + outColBase + col] =
              (bf16)(m ? hNew : 0.0f);
        }
      }
    }
    __syncthreads();   // all WMMA A-reads of hbuf complete
#pragma unroll
    for (int sub = 0; sub < 4; ++sub) {
      const int col = hblk + sub * 16 + nl;
#pragma unroll
      for (int v = 0; v < 8; ++v) hbuf[bIdx[v] * HDIM + col] = hkeep[sub][v];
    }
    __syncthreads();   // h(t+1) visible before next step's reads
  }
  if (hFinal) {
    for (int i = tid; i < BATCH * HDIM; i += 1024) hFinal[i] = (float)hbuf[i];
  }
}

// d_out[b][0:512] = hf_pre, d_out[b][512:1024] = hb_fol  (fp32)
__global__ void finalize_k(const float* __restrict__ hp, const float* __restrict__ hf,
                           float* __restrict__ out) {
  int i = blockIdx.x * blockDim.x + threadIdx.x;
  if (i < BATCH * 1024) {
    int b = i >> 10, j = i & 1023;
    out[i] = (j < 512) ? hp[b * 512 + j] : hf[b * 512 + (j - 512)];
  }
}

// ---------------------------------------------------------------------------
// Host launcher
// Input order: 0=embedded, then pre_params L0{wih_f,whh_f,b_f,wih_b,whh_b,b_b},
// L1{...}, fol_params L0{...}, L1{...}, 25=offsets, 26=stc_lens.
// ---------------------------------------------------------------------------
extern "C" void kernel_launch(void* const* d_in, const int* in_sizes, int n_in,
                              void* d_out, int out_size, void* d_ws, size_t ws_size,
                              hipStream_t stream) {
  (void)in_sizes; (void)n_in; (void)out_size; (void)ws_size;
  const float* emb = (const float*)d_in[0];
  const int* off = (const int*)d_in[25];
  const int* stc = (const int*)d_in[26];

  char* p = (char*)d_ws;
  auto carve = [&](size_t bytes) -> char* {
    char* r = p; p += (bytes + 255) & ~(size_t)255; return r;
  };
  bf16* xbuf[2];
  xbuf[0] = (bf16*)carve((size_t)TB * EDIM * 2);             // 16 MB (pre_x)
  xbuf[1] = (bf16*)carve((size_t)TB * EDIM * 2);             // 16 MB (fol_x)
  bf16* out0[2];
  out0[0] = (bf16*)carve((size_t)TB * 1024 * 2);             // 32 MB (pre layer0)
  out0[1] = (bf16*)carve((size_t)TB * 1024 * 2);             // 32 MB (fol layer0)
  bf16* zxb[4];
  for (int r = 0; r < 4; ++r)
    zxb[r] = (bf16*)carve((size_t)TB * G4H * 2);             // 4 x 64 MB
  int*   lensPre = (int*)  carve(BATCH * 4);
  int*   lensFol = (int*)  carve(BATCH * 4);
  float* hfin    = (float*)carve((size_t)4 * BATCH * HDIM * 4);

  bf16* wih[2][2][2]; bf16* whh[2][2][2]; const float* bia[2][2][2];
  for (int m = 0; m < 2; ++m)
    for (int l = 0; l < 2; ++l)
      for (int d = 0; d < 2; ++d) {
        const int inDim = l ? 1024 : 512;
        const size_t nih = (size_t)G4H * inDim, nhh = (size_t)G4H * HDIM;
        const int base = 1 + m * 12 + l * 6;
        const float* sih = (const float*)d_in[base + (d ? 3 : 0)];
        const float* shh = (const float*)d_in[base + (d ? 4 : 1)];
        bia[m][l][d] = (const float*)d_in[base + (d ? 5 : 2)];
        wih[m][l][d] = (bf16*)carve(nih * 2);
        whh[m][l][d] = (bf16*)carve(nhh * 2);
        cvt_k<<<1024, 256, 0, stream>>>(sih, wih[m][l][d], (int)nih);
        cvt_k<<<1024, 256, 0, stream>>>(shh, whh[m][l][d], (int)nhh);
      }
  prep_lens_k<<<1, 64, 0, stream>>>(off, stc, lensPre, lensFol);
  prep_x_k<<<2048, 256, 0, stream>>>(emb, off, stc, xbuf[0], 0);
  prep_x_k<<<2048, 256, 0, stream>>>(emb, off, stc, xbuf[1], 1);

  const dim3 gGemm(G4H / 64, TB / 128), bGemm(256);
  const int* lensArr[2] = { lensPre, lensFol };

  // ---- layer 0: 4 independent zx GEMMs, then one fused 4-WGP recurrence ----
  RecArgs a0{};
  for (int r = 0; r < 4; ++r) {
    const int m = r >> 1, d = r & 1;
    zx_gemm_k<<<gGemm, bGemm, 0, stream>>>(xbuf[m], wih[m][0][d], bia[m][0][d],
                                           zxb[r], EDIM);
    a0.zx[r] = zxb[r];        a0.whh[r] = whh[m][0][d];
    a0.lens[r] = lensArr[m];  a0.outSeq[r] = out0[m];
    a0.hFinal[r] = nullptr;   a0.outCol[r] = d ? HDIM : 0;
    a0.reverse[r] = d;
  }
  lstm_rec_k<<<4, 1024, 0, stream>>>(a0);

  // ---- layer 1: 4 independent zx GEMMs, then one fused 4-WGP recurrence ----
  RecArgs a1{};
  for (int r = 0; r < 4; ++r) {
    const int m = r >> 1, d = r & 1;
    zx_gemm_k<<<gGemm, bGemm, 0, stream>>>(out0[m], wih[m][1][d], bia[m][1][d],
                                           zxb[r], 1024);
    a1.zx[r] = zxb[r];        a1.whh[r] = whh[m][1][d];
    a1.lens[r] = lensArr[m];  a1.outSeq[r] = nullptr;
    a1.hFinal[r] = hfin + (size_t)r * BATCH * HDIM;
    a1.outCol[r] = 0;         a1.reverse[r] = d;
  }
  lstm_rec_k<<<4, 1024, 0, stream>>>(a1);

  finalize_k<<<(BATCH * 1024 + 255) / 256, 256, 0, stream>>>(
      hfin /*run0 = pre fwd*/, hfin + (size_t)3 * BATCH * HDIM /*run3 = fol bwd*/,
      (float*)d_out);
}
